// DMK_I_62216896250345
// MI455X (gfx1250) — matrix operand order repaired and checked by reading, not compiled, not run on hardware
//
#include <hip/hip_runtime.h>
#include <hip/hip_fp16.h>

// ---------------- model constants ----------------
constexpr int B_  = 16;
constexpr int L_  = 1024;
constexpr int DM  = 128;   // d_model
constexpr int DI  = 256;   // d_inner
constexpr int DS  = 16;    // d_state
constexpr int DTR = 8;     // dt_rank
constexpr int NL  = 4;     // layers

// ---------------- workspace layout (bytes) ----------------
constexpr size_t OFF_H      = 0;                                    // f32 [B,L,DM]
constexpr size_t OFF_H16    = OFF_H      + (size_t)B_*L_*DM*4;      // f16 [B,L,DM]
constexpr size_t OFF_XZ     = OFF_H16    + (size_t)B_*L_*DM*2;      // f32 [B,L,2*DI]
constexpr size_t OFF_XC     = OFF_XZ     + (size_t)B_*L_*2*DI*4;    // f32 [B,L,DI]
constexpr size_t OFF_DT     = OFF_XC     + (size_t)B_*L_*DI*4;      // f32 [B,L,DI]
constexpr size_t OFF_BC     = OFF_DT     + (size_t)B_*L_*DI*4;      // f32 [B,L,32] (Bt|Ct)
constexpr size_t OFF_YG16   = OFF_BC     + (size_t)B_*L_*32*4;      // f16 [B,L,DI]
constexpr size_t OFF_YFB    = OFF_YG16   + (size_t)B_*L_*DI*2;      // f32 [B,L,DM]
constexpr size_t OFF_INW16T = OFF_YFB    + (size_t)B_*L_*DM*4;      // f16 [8][512][128] (N-major)
constexpr size_t OFF_OUTW16T= OFF_INW16T + (size_t)8*DM*2*DI*2;     // f16 [8][128][256] (N-major)
constexpr size_t OFF_POOL   = OFF_OUTW16T+ (size_t)8*DI*DM*2;       // f32 [B,DM]

// ---------------- types ----------------
typedef __attribute__((ext_vector_type(16))) _Float16 v16h;
typedef __attribute__((ext_vector_type(8)))  _Float16 v8h;
typedef __attribute__((ext_vector_type(8)))  float    v8f;

__device__ __forceinline__ float dsigmoid(float x){ return 1.f/(1.f+__expf(-x)); }
__device__ __forceinline__ float dsilu(float x){ return x*dsigmoid(x); }
__device__ __forceinline__ float dsoftplus(float x){ return x>20.f ? x : __logf(1.f+__expf(x)); }

// A operand: contiguous 16-half run per lane (matches B k order)
__device__ __forceinline__ v16h lda16(const _Float16* p){
    v8h lo = *(const v8h*)p;
    v8h hi = *(const v8h*)(p + 8);
    return __builtin_shufflevector(lo, hi, 0,1,2,3,4,5,6,7,8,9,10,11,12,13,14,15);
}
// B operand from N-major (transposed) weights: contiguous 16-half run per lane
__device__ __forceinline__ v16h ldb16(const _Float16* p){
    v8h lo = *(const v8h*)p;
    v8h hi = *(const v8h*)(p + 8);
    return __builtin_shufflevector(lo, hi, 0,1,2,3,4,5,6,7,8,9,10,11,12,13,14,15);
}

// ---------------- front-end: x -> h[B,L,DM] ----------------
__global__ void k_frontend(const float* __restrict__ x,
                           const float* __restrict__ dww, const float* __restrict__ dwb,
                           const float* __restrict__ pww, const float* __restrict__ pwb,
                           const float* __restrict__ gww, const float* __restrict__ gwb,
                           const float* __restrict__ projw, const float* __restrict__ projb,
                           float* __restrict__ h)
{
    int gid = blockIdx.x*blockDim.x + threadIdx.x;
    if (gid >= B_*L_*DM) return;
    int d = gid % DM;
    int t = (gid/DM) % L_;
    int b = gid/(DM*L_);
    const float* xb = x + (size_t)b*L_;

    float feat[2];
    #pragma unroll
    for (int c=0;c<2;c++){
        float acc = dwb[c];
        #pragma unroll
        for (int k=0;k<3;k++){
            int p = t-1+k;
            if (p<0 || p>=L_) continue;
            float v = (c==0) ? xb[p] : ((p==0)?0.f:(xb[p]-xb[p-1]));
            acc += v * dww[c*3+k];
        }
        feat[c]=acc;
    }
    float f2[2], hh[2];
    #pragma unroll
    for (int o=0;o<2;o++) f2[o] = feat[0]*pww[o*2+0] + feat[1]*pww[o*2+1] + pwb[o];
    #pragma unroll
    for (int o=0;o<2;o++){
        float g = dsigmoid(f2[0]*gww[o*2+0] + f2[1]*gww[o*2+1] + gwb[o]);
        hh[o] = f2[o]*g;
    }
    h[gid] = hh[0]*projw[d] + hh[1]*projw[DM+d] + projb[d];
}

// ------- weight convert + transpose: out[mat][n*K+k] = (f16) in[mat][k*N+n] -------
__global__ void k_cvtT(const float* __restrict__ in, _Float16* __restrict__ out,
                       int K, int N, int nmat)
{
    int idx = blockIdx.x*blockDim.x + threadIdx.x;
    if (idx >= nmat*K*N) return;
    int mat = idx / (K*N);
    int rem = idx % (K*N);
    int k = rem / N, n = rem % N;
    out[(size_t)mat*K*N + (size_t)n*K + k] = (_Float16)in[idx];
}

// ---------------- h -> h16 with optional time flip ----------------
__global__ void k_h16(const float* __restrict__ h, _Float16* __restrict__ h16, int flip){
    int gid = blockIdx.x*blockDim.x + threadIdx.x;
    if (gid >= B_*L_*DM) return;
    int d = gid % DM;
    int t = (gid/DM) % L_;
    int b = gid/(DM*L_);
    int ts = flip ? (L_-1-t) : t;
    h16[gid] = (_Float16)h[((size_t)b*L_+ts)*DM + d];
}

// ---------------- WMMA GEMM: out[M,N] = A[M,K] x WT^T, WT is [N,K] ----------------
// One wave computes a 16x64 strip (4 adjacent 16x16 N-tiles, A reused 4x).
// Software-pipelined: chunk c+1 loads are issued before chunk c WMMAs consume.
// mode 0: out = result       mode 1: out[flip_t(row)] += result
template<int K>
__global__ void k_gemm_wmma(const _Float16* __restrict__ A, const _Float16* __restrict__ WT,
                            float* __restrict__ out, int M, int N, int mode)
{
    int wave = (blockIdx.x*blockDim.x + threadIdx.x) >> 5;
    int lane = threadIdx.x & 31;
    int stripsN = N >> 6;                  // N/64
    if (wave >= (M>>4)*stripsN) return;
    int tm = wave / stripsN, ts = wave % stripsN;

    int m    = tm*16 + (lane & 15);
    int ko_a = (lane >> 4) * 16;           // A K-base (contiguous, same as B)
    int ko_b = (lane >> 4) * 16;           // B K-base (contiguous, same as A)
    int n    = ts*64 + (lane & 15);

    const _Float16* aRow = A  + (size_t)m*K + ko_a;
    const _Float16* w0   = WT + (size_t)(n     )*K + ko_b;
    const _Float16* w1   = WT + (size_t)(n + 16)*K + ko_b;
    const _Float16* w2   = WT + (size_t)(n + 32)*K + ko_b;
    const _Float16* w3   = WT + (size_t)(n + 48)*K + ko_b;

    constexpr int NCH = K/32;              // 4 or 8 chunks
    v8f c0 = {}, c1 = {}, c2 = {}, c3 = {};

    // prologue: chunk 0 in flight
    v16h a  = lda16(aRow);
    v16h b0 = ldb16(w0);
    v16h b1 = ldb16(w1);
    v16h b2 = ldb16(w2);
    v16h b3 = ldb16(w3);

    #pragma unroll
    for (int c = 0; c < NCH; c++){
        v16h an, b0n, b1n, b2n, b3n;
        if (c + 1 < NCH){
            int kn = (c+1)*32;
            __builtin_prefetch(aRow + kn + 64, 0, 1);   // global_prefetch_b8
            an  = lda16(aRow + kn);                     // issue next-chunk loads first
            b0n = ldb16(w0 + kn);
            b1n = ldb16(w1 + kn);
            b2n = ldb16(w2 + kn);
            b3n = ldb16(w3 + kn);
        }
        c0 = __builtin_amdgcn_wmma_f32_16x16x32_f16(false, a, false, b0, (short)0, c0, false, false);
        c1 = __builtin_amdgcn_wmma_f32_16x16x32_f16(false, a, false, b1, (short)0, c1, false, false);
        c2 = __builtin_amdgcn_wmma_f32_16x16x32_f16(false, a, false, b2, (short)0, c2, false, false);
        c3 = __builtin_amdgcn_wmma_f32_16x16x32_f16(false, a, false, b3, (short)0, c3, false, false);
        if (c + 1 < NCH){
            a = an; b0 = b0n; b1 = b1n; b2 = b2n; b3 = b3n;
        }
    }

    int rowBase = tm*16 + 8*(lane>>4);
    int colBase = ts*64 + (lane & 15);
    v8f acc[4] = {c0, c1, c2, c3};
    #pragma unroll
    for (int j=0;j<4;j++){
        int col = colBase + j*16;
        #pragma unroll
        for (int r=0;r<8;r++){
            int row = rowBase + r;
            if (mode == 0){
                out[(size_t)row*N + col] = acc[j][r];
            } else {
                int b = row / L_, t = row % L_;
                size_t orow = (size_t)b*L_ + (L_-1-t);
                out[orow*N + col] += acc[j][r];
            }
        }
    }
}

// ------- causal depthwise conv + SiLU + x_proj + dt(softplus); one block per token -------
__global__ void k_conv_xproj(const float* __restrict__ xz,
                             const float* __restrict__ convw, const float* __restrict__ convb,
                             const float* __restrict__ xprojw,
                             const float* __restrict__ dtw, const float* __restrict__ dtb,
                             float* __restrict__ xc, float* __restrict__ dt,
                             float* __restrict__ bc)
{
    __shared__ float s_xc[DI];
    __shared__ float s_proj[DTR];
    int bt = blockIdx.x;          // b*L + t
    int t  = bt % L_;
    int d  = threadIdx.x;         // 0..255

    float acc = convb[d];
    #pragma unroll
    for (int k=0;k<4;k++){
        int p = t-3+k;
        if (p>=0) acc += xz[(size_t)(bt-3+k)*(2*DI) + d] * convw[d*4+k];
    }
    float v = dsilu(acc);
    s_xc[d] = v;
    xc[(size_t)bt*DI + d] = v;
    __syncthreads();

    if (d < DTR + 2*DS) {         // 40 projection outputs
        float p = 0.f;
        for (int i=0;i<DI;i++) p += s_xc[i]*xprojw[i*(DTR+2*DS) + d];
        if (d < DTR) s_proj[d] = p;
        else         bc[(size_t)bt*32 + (d-DTR)] = p;   // [0..15]=Bt, [16..31]=Ct
    }
    __syncthreads();

    float pv = dtb[d];
    #pragma unroll
    for (int r=0;r<DTR;r++) pv += s_proj[r]*dtw[r*DI + d];
    dt[(size_t)bt*DI + d] = dsoftplus(pv);
}

// ------- selective scan: one lane per (b, d_inner); gating fused, emits f16 -------
__global__ void k_scan(const float* __restrict__ dt, const float* __restrict__ xc,
                       const float* __restrict__ bc, const float* __restrict__ xz,
                       const float* __restrict__ Alog, const float* __restrict__ Dp,
                       _Float16* __restrict__ yg16)
{
    int gid = blockIdx.x*blockDim.x + threadIdx.x;
    if (gid >= B_*DI) return;
    int d = gid % DI, b = gid / DI;

    float A[DS], hs[DS];
    #pragma unroll
    for (int n=0;n<DS;n++){ A[n] = -__expf(Alog[d*DS+n]); hs[n]=0.f; }
    float Dv = Dp[d];

    for (int t=0;t<L_;t++){
        size_t bt = (size_t)b*L_ + t;
        float dtv = dt[bt*DI+d];
        float xv  = xc[bt*DI+d];
        float zv  = xz[bt*2*DI + DI + d];
        float dbx = dtv*xv;
        float y = 0.f;
        #pragma unroll
        for (int n=0;n<DS;n++){
            float Bn = bc[bt*32+n];
            float Cn = bc[bt*32+DS+n];
            hs[n] = __expf(dtv*A[n])*hs[n] + dbx*Bn;
            y += hs[n]*Cn;
        }
        float yg = (y + Dv*xv) * dsilu(zv);
        yg16[bt*DI+d] = (_Float16)yg;
    }
}

// ------- h += layernorm(yf+yb); one block (128 thr) per token -------
__global__ void k_ln_res(const float* __restrict__ yfb, const float* __restrict__ g,
                         const float* __restrict__ bb, float* __restrict__ h)
{
    __shared__ float red[DM];
    int row = blockIdx.x, d = threadIdx.x;
    float v = yfb[(size_t)row*DM + d];
    red[d] = v; __syncthreads();
    #pragma unroll
    for (int s=DM/2;s>0;s>>=1){ if (d<s) red[d]+=red[d+s]; __syncthreads(); }
    float mu = red[0]/DM; __syncthreads();
    float dv = v-mu;
    red[d]=dv*dv; __syncthreads();
    #pragma unroll
    for (int s=DM/2;s>0;s>>=1){ if (d<s) red[d]+=red[d+s]; __syncthreads(); }
    float var = red[0]/DM;
    h[(size_t)row*DM+d] += dv*rsqrtf(var+1e-5f)*g[d] + bb[d];
}

// ------- mean over L -------
__global__ void k_pool(const float* __restrict__ h, float* __restrict__ pooled){
    int gid = blockIdx.x*blockDim.x + threadIdx.x;
    if (gid >= B_*DM) return;
    int d = gid % DM, b = gid / DM;
    float s = 0.f;
    for (int t=0;t<L_;t++) s += h[((size_t)b*L_+t)*DM+d];
    pooled[gid] = s/(float)L_;
}

// ------- KAN head: B-splines (order 3, grid 5) + silu base, sigmoid out -------
__global__ void k_kan(const float* __restrict__ pooled, const float* __restrict__ basew,
                      const float* __restrict__ splw, const float* __restrict__ scaler,
                      float* __restrict__ out)
{
    int b = threadIdx.x;
    if (b >= B_) return;
    float g[12];
    #pragma unroll
    for (int j=0;j<12;j++) g[j] = -1.f + 0.4f*(float)(j-3);
    float acc = 0.f;
    for (int d=0; d<DM; d++){
        float p = pooled[b*DM+d];
        acc += dsilu(p)*basew[d];
        float bas[11];
        #pragma unroll
        for (int j=0;j<11;j++) bas[j] = (p>=g[j] && p<g[j+1]) ? 1.f : 0.f;
        #pragma unroll
        for (int k=1;k<=3;k++){
            for (int j=0;j<11-k;j++){
                float left  = (p-g[j])/(g[j+k]-g[j])*bas[j];
                float right = (g[j+k+1]-p)/(g[j+k+1]-g[j+1])*bas[j+1];
                bas[j] = left+right;
            }
        }
        float sc = scaler[d];
        #pragma unroll
        for (int k=0;k<8;k++) acc += bas[k]*splw[d*8+k]*sc;
    }
    out[b] = dsigmoid(acc);
}

// ---------------- host orchestration ----------------
extern "C" void kernel_launch(void* const* d_in, const int* in_sizes, int n_in,
                              void* d_out, int out_size, void* d_ws, size_t ws_size,
                              hipStream_t stream) {
    const float* x        = (const float*)d_in[0];
    const float* tff_dw_w = (const float*)d_in[1];
    const float* tff_dw_b = (const float*)d_in[2];
    const float* tff_pw_w = (const float*)d_in[3];
    const float* tff_pw_b = (const float*)d_in[4];
    const float* tff_gw   = (const float*)d_in[5];
    const float* tff_gb   = (const float*)d_in[6];
    const float* proj_w   = (const float*)d_in[7];
    const float* proj_b   = (const float*)d_in[8];
    const float* m_in_w   = (const float*)d_in[9];
    const float* m_conv_w = (const float*)d_in[10];
    const float* m_conv_b = (const float*)d_in[11];
    const float* m_xproj  = (const float*)d_in[12];
    const float* m_dt_w   = (const float*)d_in[13];
    const float* m_dt_b   = (const float*)d_in[14];
    const float* m_Alog   = (const float*)d_in[15];
    const float* m_D      = (const float*)d_in[16];
    const float* m_out_w  = (const float*)d_in[17];
    const float* ln_g     = (const float*)d_in[18];
    const float* ln_b     = (const float*)d_in[19];
    const float* kan_bw   = (const float*)d_in[20];
    const float* kan_sw   = (const float*)d_in[21];
    const float* kan_sc   = (const float*)d_in[22];

    char* ws = (char*)d_ws;
    float*    H       = (float*)   (ws + OFF_H);
    _Float16* H16     = (_Float16*)(ws + OFF_H16);
    float*    XZ      = (float*)   (ws + OFF_XZ);
    float*    XC      = (float*)   (ws + OFF_XC);
    float*    DT      = (float*)   (ws + OFF_DT);
    float*    BC      = (float*)   (ws + OFF_BC);
    _Float16* YG16    = (_Float16*)(ws + OFF_YG16);
    float*    YFB     = (float*)   (ws + OFF_YFB);
    _Float16* INW16T  = (_Float16*)(ws + OFF_INW16T);
    _Float16* OUTW16T = (_Float16*)(ws + OFF_OUTW16T);
    float*    POOL    = (float*)   (ws + OFF_POOL);

    // weight f32->f16 with transpose to N-major (all 8 layer-dirs at once)
    {
        int n1 = 8*DM*2*DI, n2 = 8*DI*DM;
        k_cvtT<<<(n1+255)/256,256,0,stream>>>(m_in_w,  INW16T,  DM, 2*DI, 8);
        k_cvtT<<<(n2+255)/256,256,0,stream>>>(m_out_w, OUTW16T, DI, DM,   8);
    }

    // front-end
    {
        int n = B_*L_*DM;
        k_frontend<<<(n+255)/256,256,0,stream>>>(x, tff_dw_w, tff_dw_b, tff_pw_w, tff_pw_b,
                                                 tff_gw, tff_gb, proj_w, proj_b, H);
    }

    const int M = B_*L_;
    for (int layer=0; layer<NL; layer++){
        for (int dir=0; dir<2; dir++){
            int ld = 2*layer + dir;
            int n = B_*L_*DM;
            k_h16<<<(n+255)/256,256,0,stream>>>(H, H16, dir);

            // in_proj: [M,DM] x [DM, 2*DI], 16x64 strip per wave
            {
                int waves = (M/16)*((2*DI)/64);
                k_gemm_wmma<DM><<<waves/4,128,0,stream>>>(H16, INW16T + (size_t)ld*DM*2*DI,
                                                          XZ, M, 2*DI, 0);
            }

            k_conv_xproj<<<M,DI,0,stream>>>(XZ,
                                            m_conv_w + (size_t)ld*DI*4,
                                            m_conv_b + (size_t)ld*DI,
                                            m_xproj  + (size_t)ld*DI*(DTR+2*DS),
                                            m_dt_w   + (size_t)ld*DTR*DI,
                                            m_dt_b   + (size_t)ld*DI,
                                            XC, DT, BC);

            k_scan<<<(B_*DI+255)/256,256,0,stream>>>(DT, XC, BC, XZ,
                                                     m_Alog + (size_t)ld*DI*DS,
                                                     m_D    + (size_t)ld*DI,
                                                     YG16);

            // out_proj: [M,DI] x [DI,DM]; dir==1 flips rows and accumulates
            {
                int waves = (M/16)*(DM/64);
                k_gemm_wmma<DI><<<waves/4,128,0,stream>>>(YG16, OUTW16T + (size_t)ld*DI*DM,
                                                          YFB, M, DM, dir);
            }
        }
        k_ln_res<<<M,DM,0,stream>>>(YFB, ln_g + (size_t)layer*DM, ln_b + (size_t)layer*DM, H);
    }

    k_pool<<<(B_*DM+255)/256,256,0,stream>>>(H, POOL);
    k_kan<<<1,32,0,stream>>>(POOL, kan_bw, kan_sw, kan_sc, (float*)d_out);
}
